// GeometricSelfAttention_33174327394474
// MI455X (gfx1250) — matrix operand order repaired and checked by reading, not compiled, not run on hardware
//
#include <hip/hip_runtime.h>

// ---------------------------------------------------------------------------
// Geometric self-attention (IPA-style) for MI455X / gfx1250.
// All matmuls use V_WMMA_F32_16X16X32_F16 (f16 in, f32 accumulate), wave32.
// Flash-attention streaming: the S x S logits never touch HBM.
// V / vp are stored transposed so every WMMA B-operand fragment is contiguous
// along K and lowers to global_load_b128 pairs.
// ---------------------------------------------------------------------------

#define DEVINL __device__ __forceinline__

typedef _Float16 v16h __attribute__((ext_vector_type(16)));
typedef float    v8f  __attribute__((ext_vector_type(8)));

union AF16 { v16h v; _Float16 h[16]; };
union CF32 { v8f  v; float    f[8];  };

// Problem constants
constexpr int kB = 2, kS = 1024, kD = 512, kH = 8, kP = 16, kDH = 64;
constexpr int kHP3 = 384;          // H*P*3
constexpr int kCT  = 2688;         // 3*D + 3*HP3 (concatenated projection cols)
constexpr int kCC  = 896;          // D + HP3 (concat af|ap)
constexpr float kCUT = 15.0f, kSCALE = 0.125f; // DH^-0.5

DEVINL v8f wmma_f16(const AF16& a, const AF16& b, v8f c) {
  return __builtin_amdgcn_wmma_f32_16x16x32_f16(false, a.v, false, b.v,
                                                (short)0, c, false, false);
}

// --- A operand (16x32 f16, M x K), row-major source, row stride in elems.
// Layout (ISA 7.12.2): lanes 0-15 K=0..7 & 16..23; lanes 16-31 K=8..15 & 24..31.
DEVINL AF16 loadA16(const _Float16* base, int rowStride) {
  int lid = threadIdx.x & 31, m = lid & 15, hf = lid >> 4;
  const _Float16* p = base + (long)m * rowStride;
  AF16 a;
#pragma unroll
  for (int j = 0; j < 8; ++j) {
    int k = hf * 8 + ((j >= 4) ? 16 : 0) + (j & 3) * 2;
    a.h[2 * j] = p[k]; a.h[2 * j + 1] = p[k + 1];
  }
  return a;
}

// Same pattern but from a 16x32 f32 LDS tile (probabilities -> f16 A operand).
DEVINL AF16 loadA_lds(const float* tile) {
  int lid = threadIdx.x & 31, m = lid & 15, hf = lid >> 4;
  AF16 a;
#pragma unroll
  for (int j = 0; j < 8; ++j) {
    int k = hf * 8 + ((j >= 4) ? 16 : 0) + (j & 3) * 2;
    a.h[2 * j]     = (_Float16)tile[m * 32 + k];
    a.h[2 * j + 1] = (_Float16)tile[m * 32 + k + 1];
  }
  return a;
}

// --- B operand (32x16 f16, K x N). Layout: lanes 0-15 K=0..7 & 16..23, lanes 16-31
// K=8..15 & 24..31 (same as A). Column-major source: element (k,n) at
// base + n*colStride + k (K contiguous per column -> b128 loads).
DEVINL AF16 loadBc(const _Float16* base, long colStride) {
  int lid = threadIdx.x & 31, n = lid & 15, kh = lid >> 4;
  const _Float16* p = base + (long)n * colStride + kh * 8;
  AF16 b;
#pragma unroll
  for (int j = 0; j < 8; ++j) { int k = ((j >= 4) ? 16 : 0) + (j & 3) * 2; b.h[2 * j] = p[k]; b.h[2 * j + 1] = p[k + 1]; }
  return b;
}

// ---------------------------------------------------------------------------
// 1) LayerNorm(x) -> f16 xn
// ---------------------------------------------------------------------------
__global__ void k_ln1(const float* x, const float* g, const float* bb, _Float16* xnb) {
  __shared__ float r1[256], r2[256];
  int t = threadIdx.x; long row = blockIdx.x;
  const float* xr = x + row * kD;
  float s = 0.f, s2 = 0.f;
  for (int i = t; i < kD; i += 256) { float v = xr[i]; s += v; s2 += v * v; }
  r1[t] = s; r2[t] = s2; __syncthreads();
  for (int o = 128; o > 0; o >>= 1) {
    if (t < o) { r1[t] += r1[t + o]; r2[t] += r2[t + o]; }
    __syncthreads();
  }
  float mean = r1[0] / (float)kD;
  float var  = r2[0] / (float)kD - mean * mean;
  float rstd = rsqrtf(var + 1e-5f);
  for (int i = t; i < kD; i += 256)
    xnb[row * kD + i] = (_Float16)((xr[i] - mean) * rstd * g[i] + bb[i]);
}

// ---------------------------------------------------------------------------
// 2) Weight transpose/convert: Wt[c][d] f16 (c = concat col, d = K dim)
// ---------------------------------------------------------------------------
__global__ void k_convW(const float* Wq, const float* Wk, const float* Wv,
                        const float* Wqp, const float* Wkp, const float* Wvp,
                        _Float16* Wt) {
  int idx = blockIdx.x * 256 + threadIdx.x;  // kCT * kD
  int c = idx >> 9, d = idx & 511;
  float v;
  if (c < 512)        v = Wq[d * 512 + c];
  else if (c < 1024)  v = Wk[d * 512 + (c - 512)];
  else if (c < 1536)  v = Wv[d * 512 + (c - 1024)];
  else {
    int cc = c - 1536;
    if (cc < 384)      v = Wqp[d * 384 + cc];
    else if (cc < 768) v = Wkp[d * 384 + (cc - 384)];
    else               v = Wvp[d * 384 + (cc - 768)];
  }
  Wt[(long)c * 512 + d] = (_Float16)v;
}

__global__ void k_convWo(const float* Wo, _Float16* WoT) {
  int idx = blockIdx.x * 256 + threadIdx.x;  // kD * kCC
  int n = idx / kCC, k = idx % kCC;
  WoT[(long)n * kCC + k] = (_Float16)Wo[(long)k * kD + n];
}

// ---------------------------------------------------------------------------
// 3) Fused projection GEMM: proj[2048][2688] = xn @ [Wq|Wk|Wv|Wqp|Wkp|Wvp]+bias
// ---------------------------------------------------------------------------
__global__ void __launch_bounds__(256) k_proj(const _Float16* xnb, const _Float16* Wt,
                                              const float* bqp, const float* bkp,
                                              const float* bvp, float* proj) {
  int wv = threadIdx.x >> 5;
  int unit = blockIdx.x * 8 + wv;          // 128 * 168 tiles
  int mt = unit / 168, nt = unit % 168;
  int lid = threadIdx.x & 31, n = lid & 15, hf = lid >> 4;
  CF32 c;
#pragma unroll
  for (int r = 0; r < 8; ++r) c.f[r] = 0.f;
  for (int kc = 0; kc < 16; ++kc) {
    AF16 a = loadA16(xnb + (long)(mt * 16) * kD + kc * 32, kD);
    AF16 b = loadBc(Wt + (long)(nt * 16) * kD + kc * 32, kD);
    c.v = wmma_f16(a, b, c.v);
  }
  int col = nt * 16 + n;
  float bias = 0.f;
  if (col >= 2304)      bias = bvp[col - 2304];
  else if (col >= 1920) bias = bkp[col - 1920];
  else if (col >= 1536) bias = bqp[col - 1536];
#pragma unroll
  for (int r = 0; r < 8; ++r)
    proj[(long)(mt * 16 + hf * 8 + r) * kCT + col] = c.f[r] + bias;
}

// ---------------------------------------------------------------------------
// 4) Prep: head-major f16 planes.
//    Qb/Kb/QPb/KPb : [b][h][s][64] (A / K-side B operands, K contiguous)
//    VbT           : [b][h][64][s] (transposed so PV B operand is contiguous)
//    VPbT          : [b][h][48][s]
//    QPb/KPb get +coords and zero-pad 48->64.
// ---------------------------------------------------------------------------
__global__ void k_prepqkv(const float* proj, const float* coords,
                          _Float16* Qb, _Float16* Kb, _Float16* VbT,
                          _Float16* QPb, _Float16* KPb, _Float16* VPbT) {
  long tok = blockIdx.x; int b = (int)(tok >> 10), s = (int)(tok & 1023);
  const float* pr = proj + tok * kCT;
  const float* cd = coords + tok * 3;
  for (int c = threadIdx.x; c < kCT; c += 256) {
    float v = pr[c];
    if (c < 512) {
      int h = c >> 6, d = c & 63;
      Qb[((long)(b * kH + h) * kS + s) * 64 + d] = (_Float16)v;
    } else if (c < 1024) {
      int cc = c - 512; int h = cc >> 6, d = cc & 63;
      Kb[((long)(b * kH + h) * kS + s) * 64 + d] = (_Float16)v;
    } else if (c < 1536) {
      int cc = c - 1024; int h = cc >> 6, d = cc & 63;
      VbT[((long)(b * kH + h) * 64 + d) * kS + s] = (_Float16)v;   // transposed
    } else if (c < 1920) {
      int cc = c - 1536; int h = cc / 48, r = cc % 48;
      QPb[((long)(b * kH + h) * kS + s) * 64 + r] = (_Float16)(v + cd[r % 3]);
    } else if (c < 2304) {
      int cc = c - 1920; int h = cc / 48, r = cc % 48;
      KPb[((long)(b * kH + h) * kS + s) * 64 + r] = (_Float16)(v + cd[r % 3]);
    } else {
      int cc = c - 2304; int h = cc / 48, r = cc % 48;
      VPbT[((long)(b * kH + h) * 48 + r) * kS + s] = (_Float16)v;  // transposed
    }
  }
  int t = threadIdx.x;
  if (t < 128) { int h = t >> 4, j = t & 15;
    QPb[((long)(b * kH + h) * kS + s) * 64 + 48 + j] = (_Float16)0.f;
  } else { int u = t - 128; int h = u >> 4, j = u & 15;
    KPb[((long)(b * kH + h) * kS + s) * 64 + 48 + j] = (_Float16)0.f;
  }
}

__global__ void k_prepnorm(const float* proj, const float* coords,
                           float* qn2, float* kn2) {
  int idx = blockIdx.x * 256 + threadIdx.x;   // B*H*S
  int b = idx >> 13, h = (idx >> 10) & 7, s = idx & 1023;
  const float* pr = proj + (long)(b * kS + s) * kCT;
  const float* cd = coords + (long)(b * kS + s) * 3;
  float sq = 0.f, sk = 0.f;
  for (int r = 0; r < 48; ++r) {
    float ce = cd[r % 3];
    float vq = pr[1536 + h * 48 + r] + ce;
    float vk = pr[1920 + h * 48 + r] + ce;
    sq += vq * vq; sk += vk * vk;
  }
  long o = (long)(b * kH + h) * kS + s;
  qn2[o] = sq; kn2[o] = sk;
}

// ---------------------------------------------------------------------------
// 5) Flash attention: one wave per (b, h, 16-row q tile); stream K in 32s.
// ---------------------------------------------------------------------------
__global__ void __launch_bounds__(256) k_attn(
    const _Float16* Qb, const _Float16* Kb, const _Float16* VbT,
    const _Float16* QPb, const _Float16* KPb, const _Float16* VPbT,
    const float* qn2, const float* kn2, const float* coords,
    const float* Wd1, const float* bd1, const float* Wd2, const float* bd2,
    _Float16* Cc, float* APf) {
  __shared__ float sP[8][16 * 32];            // per-wave probability staging
  __shared__ float sWd1[32], sbd1[32], sWd2[256], sbd2[8];
  int t = threadIdx.x;
  if (t < 32) { sWd1[t] = Wd1[t]; sbd1[t] = bd1[t]; }
  if (t < 256) sWd2[t] = Wd2[t];
  if (t < 8)  sbd2[t] = bd2[t];
  __syncthreads();

  int wv = t >> 5, lid = t & 31, n16 = lid & 15, hf = lid >> 4;
  int unit = blockIdx.x * 8 + wv;             // B*H*(S/16) = 1024 units
  int qt = unit & 63, h = (unit >> 6) & 7, b = unit >> 9;
  int qbase = qt * 16;
  long plane = (long)(b * kH + h) * kS;
  const _Float16* Qp  = Qb   + plane * 64;
  const _Float16* Kp  = Kb   + plane * 64;
  const _Float16* Vp  = VbT  + plane * 64;   // [64][S]
  const _Float16* QPp = QPb  + plane * 64;
  const _Float16* KPp = KPb  + plane * 64;
  const _Float16* VPp = VPbT + plane * 48;   // [48][S]
  const float* qn2p = qn2 + plane;
  const float* kn2p = kn2 + plane;

  AF16 qa0  = loadA16(Qp  + (long)qbase * 64 +  0, 64);
  AF16 qa1  = loadA16(Qp  + (long)qbase * 64 + 32, 64);
  AF16 qpa0 = loadA16(QPp + (long)qbase * 64 +  0, 64);
  AF16 qpa1 = loadA16(QPp + (long)qbase * 64 + 32, 64);

  float m_[8], l_[8], qn2r[8], cqx[8], cqy[8], cqz[8];
#pragma unroll
  for (int r = 0; r < 8; ++r) {
    m_[r] = -1e30f; l_[r] = 0.f;
    int M = hf * 8 + r;
    qn2r[r] = qn2p[qbase + M];
    long ci = ((long)b * kS + qbase + M) * 3;
    cqx[r] = coords[ci]; cqy[r] = coords[ci + 1]; cqz[r] = coords[ci + 2];
  }
  CF32 caf[4], cap[3];
#pragma unroll
  for (int ti = 0; ti < 4; ++ti)
#pragma unroll
    for (int r = 0; r < 8; ++r) caf[ti].f[r] = 0.f;
#pragma unroll
  for (int ti = 0; ti < 3; ++ti)
#pragma unroll
    for (int r = 0; r < 8; ++r) cap[ti].f[r] = 0.f;

  for (int kb = 0; kb < kS; kb += 32) {
    float t0[8], t1[8];
#pragma unroll
    for (int sub = 0; sub < 2; ++sub) {
      CF32 sc, pc;
#pragma unroll
      for (int r = 0; r < 8; ++r) { sc.f[r] = 0.f; pc.f[r] = 0.f; }
      AF16 bk0 = loadBc(Kp  + (long)(kb + sub * 16) * 64 +  0, 64);
      AF16 bk1 = loadBc(Kp  + (long)(kb + sub * 16) * 64 + 32, 64);
      sc.v = wmma_f16(qa0, bk0, sc.v);
      sc.v = wmma_f16(qa1, bk1, sc.v);
      AF16 bp0 = loadBc(KPp + (long)(kb + sub * 16) * 64 +  0, 64);
      AF16 bp1 = loadBc(KPp + (long)(kb + sub * 16) * 64 + 32, 64);
      pc.v = wmma_f16(qpa0, bp0, pc.v);
      pc.v = wmma_f16(qpa1, bp1, pc.v);

      int kcol = kb + sub * 16 + n16;
      float kn2v = kn2p[kcol];
      long cki = ((long)b * kS + kcol) * 3;
      float ckx = coords[cki], cky = coords[cki + 1], ckz = coords[cki + 2];
      float* tt = sub ? t1 : t0;
#pragma unroll
      for (int r = 0; r < 8; ++r) {
        float dx = cqx[r] - ckx, dy = cqy[r] - cky, dz = cqz[r] - ckz;
        float dist = sqrtf(dx * dx + dy * dy + dz * dz);
        float db = sbd2[h];
#pragma unroll
        for (int j = 0; j < 32; ++j) {
          float hj = dist * sWd1[j] + sbd1[j];
          hj = hj > 0.f ? hj : 0.f;
          db += hj * sWd2[j * 8 + h];
        }
        float logit = sc.f[r] * kSCALE
                    - 0.5f * (qn2r[r] + kn2v - 2.0f * pc.f[r]) + db;
        tt[r] = (dist < kCUT) ? logit : -1e9f;
      }
    }
    // online softmax (rows live across 16 lanes of each half-wave)
    float alpha[8];
#pragma unroll
    for (int r = 0; r < 8; ++r) {
      float mx = fmaxf(t0[r], t1[r]);
#pragma unroll
      for (int st = 1; st < 16; st <<= 1) mx = fmaxf(mx, __shfl_xor(mx, st, 32));
      float nm = fmaxf(m_[r], mx);
      alpha[r] = __expf(m_[r] - nm);
      m_[r] = nm;
      float p0 = __expf(t0[r] - nm), p1 = __expf(t1[r] - nm);
      float rs = p0 + p1;
#pragma unroll
      for (int st = 1; st < 16; st <<= 1) rs += __shfl_xor(rs, st, 32);
      l_[r] = l_[r] * alpha[r] + rs;
      int M = hf * 8 + r;
      sP[wv][M * 32 + n16]      = p0;
      sP[wv][M * 32 + 16 + n16] = p1;
    }
#pragma unroll
    for (int ti = 0; ti < 4; ++ti)
#pragma unroll
      for (int r = 0; r < 8; ++r) caf[ti].f[r] *= alpha[r];
#pragma unroll
    for (int ti = 0; ti < 3; ++ti)
#pragma unroll
      for (int r = 0; r < 8; ++r) cap[ti].f[r] *= alpha[r];

    AF16 pa = loadA_lds(&sP[wv][0]);          // C-layout -> A-layout, f32->f16
#pragma unroll
    for (int ti = 0; ti < 4; ++ti) {
      AF16 bv = loadBc(Vp + (long)(ti * 16) * kS + kb, kS);     // contiguous K
      caf[ti].v = wmma_f16(pa, bv, caf[ti].v);
    }
#pragma unroll
    for (int ti = 0; ti < 3; ++ti) {
      AF16 bv = loadBc(VPp + (long)(ti * 16) * kS + kb, kS);    // contiguous K
      cap[ti].v = wmma_f16(pa, bv, cap[ti].v);
    }
    if (kb + 32 < kS) {
      __builtin_prefetch(Kp + (long)(kb + 32) * 64, 0, 1);
      __builtin_prefetch(KPp + (long)(kb + 32) * 64, 0, 1);
    }
  }

  // normalize and emit concat(af|ap) in f16 + ap in f32 for the coord update
#pragma unroll
  for (int r = 0; r < 8; ++r) {
    float inv = 1.0f / l_[r];
    int M = hf * 8 + r;
    long row = (long)b * kS + qbase + M;
#pragma unroll
    for (int ti = 0; ti < 4; ++ti)
      Cc[row * kCC + h * 64 + ti * 16 + n16] = (_Float16)(caf[ti].f[r] * inv);
#pragma unroll
    for (int ti = 0; ti < 3; ++ti) {
      float v = cap[ti].f[r] * inv;
      Cc[row * kCC + kD + h * 48 + ti * 16 + n16] = (_Float16)v;
      APf[(row * kH + h) * 48 + ti * 16 + n16] = v;
    }
  }
}

// ---------------------------------------------------------------------------
// 6) Output GEMM: pre[2048][512] = concat(af|ap) @ Wo + bo
// ---------------------------------------------------------------------------
__global__ void __launch_bounds__(256) k_outgemm(const _Float16* Cc, const _Float16* WoT,
                                                 const float* bo, float* pre) {
  int wv = threadIdx.x >> 5;
  int unit = blockIdx.x * 8 + wv;          // 128 * 32 tiles
  int mt = unit >> 5, nt = unit & 31;
  int lid = threadIdx.x & 31, n = lid & 15, hf = lid >> 4;
  CF32 c;
#pragma unroll
  for (int r = 0; r < 8; ++r) c.f[r] = 0.f;
  for (int kc = 0; kc < 28; ++kc) {
    AF16 a = loadA16(Cc + (long)(mt * 16) * kCC + kc * 32, kCC);
    AF16 b = loadBc(WoT + (long)(nt * 16) * kCC + kc * 32, kCC);
    c.v = wmma_f16(a, b, c.v);
  }
  int col = nt * 16 + n;
  float bias = bo[col];
#pragma unroll
  for (int r = 0; r < 8; ++r)
    pre[(long)(mt * 16 + hf * 8 + r) * kD + col] = c.f[r] + bias;
}

// ---------------------------------------------------------------------------
// 7) out = LN(residual + pre) ; coords update
// ---------------------------------------------------------------------------
__global__ void k_ln2(const float* x, const float* pre, const float* g,
                      const float* bb, float* out) {
  __shared__ float r1[256], r2[256];
  int t = threadIdx.x; long row = blockIdx.x;
  const float* xr = x + row * kD;
  const float* pr = pre + row * kD;
  float s = 0.f, s2 = 0.f;
  for (int i = t; i < kD; i += 256) { float v = xr[i] + pr[i]; s += v; s2 += v * v; }
  r1[t] = s; r2[t] = s2; __syncthreads();
  for (int o = 128; o > 0; o >>= 1) {
    if (t < o) { r1[t] += r1[t + o]; r2[t] += r2[t + o]; }
    __syncthreads();
  }
  float mean = r1[0] / (float)kD;
  float var  = r2[0] / (float)kD - mean * mean;
  float rstd = rsqrtf(var + 1e-5f);
  for (int i = t; i < kD; i += 256) {
    float v = xr[i] + pr[i];
    out[row * kD + i] = (v - mean) * rstd * g[i] + bb[i];
  }
}

__global__ void k_coords(const float* APf, const float* coords, float* outc) {
  int idx = blockIdx.x * 256 + threadIdx.x;   // B*S*3
  int tok = idx / 3, e = idx % 3;
  float s = 0.f;
  for (int h = 0; h < kH; ++h)
    for (int p = 0; p < kP; ++p)
      s += APf[((long)tok * kH + h) * 48 + p * 3 + e];
  outc[idx] = coords[idx] + s * (0.1f / 128.0f);
}

// ---------------------------------------------------------------------------
extern "C" void kernel_launch(void* const* d_in, const int* in_sizes, int n_in,
                              void* d_out, int out_size, void* d_ws, size_t ws_size,
                              hipStream_t stream) {
  const float* x      = (const float*)d_in[0];
  const float* coords = (const float*)d_in[1];
  const float* Wq  = (const float*)d_in[2];
  const float* Wk  = (const float*)d_in[3];
  const float* Wv  = (const float*)d_in[4];
  const float* Wqp = (const float*)d_in[5];
  const float* bqp = (const float*)d_in[6];
  const float* Wkp = (const float*)d_in[7];
  const float* bkp = (const float*)d_in[8];
  const float* Wvp = (const float*)d_in[9];
  const float* bvp = (const float*)d_in[10];
  const float* Wd1 = (const float*)d_in[11];
  const float* bd1 = (const float*)d_in[12];
  const float* Wd2 = (const float*)d_in[13];
  const float* bd2 = (const float*)d_in[14];
  const float* Wo  = (const float*)d_in[15];
  const float* bo  = (const float*)d_in[16];
  const float* g1  = (const float*)d_in[17];
  const float* b1  = (const float*)d_in[18];
  const float* g2  = (const float*)d_in[19];
  const float* b2  = (const float*)d_in[20];

  const long NTOK = (long)kB * kS;            // 2048
  char* w = (char*)d_ws;
  size_t off = 0;
  auto carve = [&](size_t bytes) -> char* {
    char* p = w + off;
    off = (off + bytes + 255) & ~(size_t)255;
    return p;
  };
  _Float16* xnb  = (_Float16*)carve(NTOK * kD * 2);
  _Float16* Wt   = (_Float16*)carve((size_t)kCT * kD * 2);
  _Float16* WoT  = (_Float16*)carve((size_t)kD * kCC * 2);
  float*    proj = (float*)carve(NTOK * kCT * 4);
  _Float16* Qb   = (_Float16*)carve((size_t)kB * kH * kS * 64 * 2);
  _Float16* Kb   = (_Float16*)carve((size_t)kB * kH * kS * 64 * 2);
  _Float16* VbT  = (_Float16*)carve((size_t)kB * kH * kS * 64 * 2);
  _Float16* QPb  = (_Float16*)carve((size_t)kB * kH * kS * 64 * 2);
  _Float16* KPb  = (_Float16*)carve((size_t)kB * kH * kS * 64 * 2);
  _Float16* VPbT = (_Float16*)carve((size_t)kB * kH * kS * 48 * 2);
  float*    qn2  = (float*)carve((size_t)kB * kH * kS * 4);
  float*    kn2  = (float*)carve((size_t)kB * kH * kS * 4);
  _Float16* Cc   = (_Float16*)carve(NTOK * kCC * 2);
  float*    APf  = (float*)carve(NTOK * kH * 48 * 4);
  float*    pre  = (float*)carve(NTOK * kD * 4);
  (void)ws_size; (void)n_in; (void)in_sizes; (void)out_size;  // ~51 MB used

  float* out_x = (float*)d_out;                 // [B,S,D]
  float* out_c = out_x + NTOK * kD;             // [B,S,3]

  k_ln1<<<NTOK, 256, 0, stream>>>(x, g1, b1, xnb);
  k_convW<<<(kCT * kD) / 256, 256, 0, stream>>>(Wq, Wk, Wv, Wqp, Wkp, Wvp, Wt);
  k_convWo<<<(kD * kCC) / 256, 256, 0, stream>>>(Wo, WoT);
  k_proj<<<(128 * 168) / 8, 256, 0, stream>>>(xnb, Wt, bqp, bkp, bvp, proj);
  k_prepqkv<<<NTOK, 256, 0, stream>>>(proj, coords, Qb, Kb, VbT, QPb, KPb, VPbT);
  k_prepnorm<<<(kB * kH * kS) / 256, 256, 0, stream>>>(proj, coords, qn2, kn2);
  k_attn<<<(kB * kH * (kS / 16)) / 8, 256, 0, stream>>>(
      Qb, Kb, VbT, QPb, KPb, VPbT, qn2, kn2, coords,
      Wd1, bd1, Wd2, bd2, Cc, APf);
  k_outgemm<<<(128 * 32) / 8, 256, 0, stream>>>(Cc, WoT, bo, pre);
  k_ln2<<<NTOK, 256, 0, stream>>>(x, pre, g2, b2, out_x);
  k_coords<<<(kB * kS * 3) / 256, 256, 0, stream>>>(APf, coords, out_c);
}
